// SpatiotemporalTokenizer_69930657513644
// MI455X (gfx1250) — matrix operand order repaired and checked by reading, not compiled, not run on hardware
//
#include <hip/hip_runtime.h>
#include <hip/hip_bf16.h>

// ---------------------------------------------------------------------------
// CDNA5 (gfx1250) VQ-VAE forward: bf16 WMMA implicit-GEMM convs + fused VQ.
// ---------------------------------------------------------------------------

typedef __attribute__((ext_vector_type(16))) __bf16 v16bf;
typedef __attribute__((ext_vector_type(8)))  __bf16 v8bf;
typedef __attribute__((ext_vector_type(8)))  float  v8f;

#define BB 8
#define HH 64
#define WW 64
#define CIN 16
#define HID 128
#define KCODES 8192
#define MTOT (BB*HH*WW)          // 32768 spatial positions

__device__ __forceinline__ float gelu_exact(float v) {
    return 0.5f * v * (1.0f + erff(v * 0.70710678118654752f));
}

// ---------------------------------------------------------------------------
// Packing / conversion kernels
// ---------------------------------------------------------------------------

// NCHW f32 -> NHWC bf16
__global__ __launch_bounds__(256) void pack_nchw_to_nhwc_bf16(
    const float* __restrict__ in, __bf16* __restrict__ out, int C) {
    int t = blockIdx.x * 256 + threadIdx.x;
    int total = BB * C * HH * WW;
    if (t >= total) return;
    int c = t % C;
    int x = (t / C) % WW;
    int y = (t / (C * WW)) % HH;
    int b = t / (C * WW * HH);
    out[t] = (__bf16)in[((b * C + c) * HH + y) * WW + x];
}

// w[Cout][Cin][3][3] f32 -> wt[Cout][Kpad] bf16, k = (ky*3+kx)*Cin + ci
__global__ __launch_bounds__(256) void pack_w3x3_bf16(
    const float* __restrict__ w, __bf16* __restrict__ wt,
    int Cout, int Cin, int Kpad) {
    int t = blockIdx.x * 256 + threadIdx.x;
    if (t >= Cout * Kpad) return;
    int co = t / Kpad, k = t % Kpad;
    int K = Cin * 9;
    float v = 0.0f;
    if (k < K) {
        int kk = k / Cin, ci = k % Cin;           // kk = ky*3+kx
        v = w[(co * Cin + ci) * 9 + kk];
    }
    wt[t] = (__bf16)v;
}

__global__ __launch_bounds__(256) void cast_f32_to_bf16(
    const float* __restrict__ in, __bf16* __restrict__ out, int n) {
    int t = blockIdx.x * 256 + threadIdx.x;
    if (t < n) out[t] = (__bf16)in[t];
}

__global__ __launch_bounds__(256) void embed_norm2(
    const float* __restrict__ e, float* __restrict__ n2) {
    int k = blockIdx.x * 256 + threadIdx.x;
    if (k >= KCODES) return;
    float s = 0.0f;
    for (int d = 0; d < HID; ++d) { float v = e[k * HID + d]; s += v * v; }
    n2[k] = s;
}

// ---------------------------------------------------------------------------
// 3x3 SAME conv as implicit GEMM, Cout == 128, bias + exact GELU fused.
// 256 threads = 8 waves; block tile = 16 rows x 128 cols; one 16x16 per wave.
// ---------------------------------------------------------------------------
__global__ __launch_bounds__(256) void conv3x3_wmma(
    const __bf16* __restrict__ in,   // [MTOT][Cin]  NHWC
    const __bf16* __restrict__ wt,   // [128][Kpad]
    const float*  __restrict__ bias, // [128]
    __bf16* __restrict__ out,        // [MTOT][128]  NHWC
    int Cin, int Kpad) {
    __shared__ __bf16 As[16 * 1152];               // max Kpad = 1152

    const int tid  = threadIdx.x;
    const int wave = tid >> 5;
    const int lane = tid & 31;
    const int hf   = lane >> 4;                    // half of the wave
    const int mrow = lane & 15;
    const int abase = hf * 8;
    const int m0 = blockIdx.x * 16;
    const int K  = Cin * 9;

    // build zero-padded patch tile [16][Kpad]
    for (int t = tid; t < 16 * Kpad; t += 256) {
        int mi = t / Kpad, k = t % Kpad;
        __bf16 v = (__bf16)0.0f;
        if (k < K) {
            int kk = k / Cin, ci = k % Cin;
            int ky = kk / 3, kx = kk % 3;
            int p = m0 + mi;
            int b = p / (HH * WW);
            int rem = p % (HH * WW);
            int yy = rem / WW + ky - 1;
            int xx = rem % WW + kx - 1;
            if (yy >= 0 && yy < HH && xx >= 0 && xx < WW)
                v = in[(((size_t)(b * HH + yy) * WW + xx)) * Cin + ci];
        }
        As[mi * Kpad + k] = v;
    }
    __syncthreads();

    const int co = wave * 16 + mrow;               // B column = output channel
    const __bf16* wrow = wt + (size_t)co * Kpad;

    v8f acc = {};
    for (int ks = 0; ks < Kpad; ks += 32) {
        union { v16bf v; v8bf h[2]; } a;
        a.h[0] = *(const v8bf*)&As[mrow * Kpad + ks + 2 * abase];
        a.h[1] = *(const v8bf*)&As[mrow * Kpad + ks + 2 * abase + 8];
        v16bf bfrag = *(const v16bf*)&wrow[ks + hf * 16];
        acc = __builtin_amdgcn_wmma_f32_16x16x32_bf16(
            false, a.v, false, bfrag, (short)0, acc, false, false);
    }

    const float bv = bias[co];
    for (int r = 0; r < 8; ++r) {
        int m = m0 + r + hf * 8;
        float v = gelu_exact(acc[r] + bv);
        out[(size_t)m * HID + co] = (__bf16)v;
    }
}

// ---------------------------------------------------------------------------
// Fused VQ: argmax(f.e - 0.5*|e|^2) over 8192 codes, then gather code rows.
// Block: 16 flat rows; 8 waves stripe the codebook 16 codes at a time.
// ---------------------------------------------------------------------------
__global__ __launch_bounds__(256) void vq_argmin_gather(
    const __bf16* __restrict__ h,   // [MTOT][128]
    const __bf16* __restrict__ E,   // [8192][128]
    const float*  __restrict__ n2,  // [8192]
    __bf16* __restrict__ q) {       // [MTOT][128]
    __shared__ float sv[8][16];
    __shared__ int   si[8][16];
    __shared__ int   fidx[16];

    const int tid  = threadIdx.x;
    const int wave = tid >> 5;
    const int lane = tid & 31;
    const int hf   = lane >> 4;
    const int nrow = lane & 15;                    // A row and B column index
    const int abase = hf * 8;
    const int m0 = blockIdx.x * 16;

    // A fragments for D=128 (4 x K32), kept in registers for the whole sweep
    union { v16bf v; v8bf h2[2]; } a[4];
    {
        const __bf16* arow = h + (size_t)(m0 + nrow) * HID;
        for (int ks = 0; ks < 4; ++ks) {
            a[ks].h2[0] = *(const v8bf*)(arow + ks * 32 + 2 * abase);
            a[ks].h2[1] = *(const v8bf*)(arow + ks * 32 + 2 * abase + 8);
        }
    }

    float bestv[8]; int besti[8];
    for (int r = 0; r < 8; ++r) { bestv[r] = -3.0e38f; besti[r] = 0; }

    for (int c0 = wave * 16; c0 < KCODES; c0 += 128) {
        const int c = c0 + nrow;
        const __bf16* erow = E + (size_t)c * HID;
        if (c0 + 128 < KCODES)
            __builtin_prefetch(E + (size_t)(c + 128) * HID);
        v8f acc = {};
        for (int ks = 0; ks < 4; ++ks) {
            v16bf bfrag = *(const v16bf*)(erow + ks * 32 + hf * 16);
            acc = __builtin_amdgcn_wmma_f32_16x16x32_bf16(
                false, a[ks].v, false, bfrag, (short)0, acc, false, false);
        }
        const float nv = 0.5f * n2[c];
        for (int r = 0; r < 8; ++r) {
            float s = acc[r] - nv;                  // lane's column = code c
            if (s > bestv[r]) { bestv[r] = s; besti[r] = c; }
        }
    }

    // reduce across the 16 lanes of each half (columns of the tile)
    for (int r = 0; r < 8; ++r) {
        float v = bestv[r]; int i = besti[r];
        for (int off = 1; off < 16; off <<= 1) {
            float ov = __shfl_xor(v, off, 32);
            int   oi = __shfl_xor(i, off, 32);
            if (ov > v || (ov == v && oi < i)) { v = ov; i = oi; }
        }
        bestv[r] = v; besti[r] = i;
    }
    if (nrow == 0) {                                // lanes 0 and 16
        for (int r = 0; r < 8; ++r) {
            sv[wave][hf * 8 + r] = bestv[r];
            si[wave][hf * 8 + r] = besti[r];
        }
    }
    __syncthreads();
    if (tid < 16) {
        float v = sv[0][tid]; int i = si[0][tid];
        for (int w = 1; w < 8; ++w) {
            float ov = sv[w][tid]; int oi = si[w][tid];
            if (ov > v || (ov == v && oi < i)) { v = ov; i = oi; }
        }
        fidx[tid] = i;
    }
    __syncthreads();

    // cooperative gather of the 16 winning code rows
    for (int t = tid; t < 16 * HID; t += 256) {
        int mi = t >> 7, d = t & 127;
        q[(size_t)(m0 + mi) * HID + d] = E[(size_t)fidx[mi] * HID + d];
    }
}

// ---------------------------------------------------------------------------
// Head: 1x1 conv (128 -> 16) + sigmoid, f32 NCHW output.
// 8 waves x 16 rows each = 128 rows per block; N = 16 = one WMMA tile.
// ---------------------------------------------------------------------------
__global__ __launch_bounds__(256) void head1x1_wmma(
    const __bf16* __restrict__ y,    // [MTOT][128]
    const __bf16* __restrict__ w4,   // [16][128]
    const float*  __restrict__ b4,   // [16]
    float* __restrict__ out) {       // [B][16][H][W]
    const int tid  = threadIdx.x;
    const int wave = tid >> 5;
    const int lane = tid & 31;
    const int hf   = lane >> 4;
    const int nrow = lane & 15;
    const int abase = hf * 8;
    const int m0 = blockIdx.x * 128 + wave * 16;

    const __bf16* arow = y + (size_t)(m0 + nrow) * HID;
    const __bf16* wrow = w4 + (size_t)nrow * HID;

    v8f acc = {};
    for (int ks = 0; ks < 4; ++ks) {
        union { v16bf v; v8bf h2[2]; } a;
        a.h2[0] = *(const v8bf*)(arow + ks * 32 + 2 * abase);
        a.h2[1] = *(const v8bf*)(arow + ks * 32 + 2 * abase + 8);
        v16bf bfrag = *(const v16bf*)(wrow + ks * 32 + hf * 16);
        acc = __builtin_amdgcn_wmma_f32_16x16x32_bf16(
            false, a.v, false, bfrag, (short)0, acc, false, false);
    }

    const float bv = b4[nrow];
    for (int r = 0; r < 8; ++r) {
        int p = m0 + r + hf * 8;
        int b = p / (HH * WW);
        int rem = p % (HH * WW);
        int yy = rem / WW, xx = rem % WW;
        float v = acc[r] + bv;
        v = 1.0f / (1.0f + __expf(-v));
        out[(((size_t)(b * CIN + nrow) * HH + yy)) * WW + xx] = v;
    }
}

// ---------------------------------------------------------------------------
// Launch
// ---------------------------------------------------------------------------
extern "C" void kernel_launch(void* const* d_in, const int* in_sizes, int n_in,
                              void* d_out, int out_size, void* d_ws, size_t ws_size,
                              hipStream_t stream) {
    const float* x     = (const float*)d_in[0];
    const float* ew1   = (const float*)d_in[1];
    const float* eb1   = (const float*)d_in[2];
    const float* ew2   = (const float*)d_in[3];
    const float* eb2   = (const float*)d_in[4];
    const float* embed = (const float*)d_in[5];
    const float* dw1   = (const float*)d_in[6];
    const float* db1   = (const float*)d_in[7];
    const float* dw2   = (const float*)d_in[8];
    const float* db2   = (const float*)d_in[9];
    float* out = (float*)d_out;

    // workspace carve-up (all chunks 256B-aligned)
    char* ws = (char*)d_ws;
    size_t off = 0;
    __bf16* X0  = (__bf16*)(ws + off); off += (size_t)MTOT * CIN * 2;       // 1 MB
    __bf16* W1t = (__bf16*)(ws + off); off += (size_t)HID * 160 * 2;        // 40 KB
    __bf16* W2t = (__bf16*)(ws + off); off += (size_t)HID * 1152 * 2;       // 288 KB
    __bf16* W3t = (__bf16*)(ws + off); off += (size_t)HID * 1152 * 2;       // 288 KB
    __bf16* W4t = (__bf16*)(ws + off); off += (size_t)CIN * HID * 2;        // 4 KB
    __bf16* E   = (__bf16*)(ws + off); off += (size_t)KCODES * HID * 2;     // 2 MB
    float*  N2  = (float*) (ws + off); off += (size_t)KCODES * 4;           // 32 KB
    __bf16* H1  = (__bf16*)(ws + off); off += (size_t)MTOT * HID * 2;       // 8 MB
    __bf16* H2  = (__bf16*)(ws + off); off += (size_t)MTOT * HID * 2;       // 8 MB
    __bf16* Q   = (__bf16*)(ws + off); off += (size_t)MTOT * HID * 2;       // 8 MB
    __bf16* Y3  = (__bf16*)(ws + off); off += (size_t)MTOT * HID * 2;       // 8 MB
    (void)off; (void)ws_size; (void)in_sizes; (void)n_in; (void)out_size;

    // ---- pack / convert -------------------------------------------------
    pack_nchw_to_nhwc_bf16<<<(MTOT * CIN + 255) / 256, 256, 0, stream>>>(x, X0, CIN);
    pack_w3x3_bf16<<<(HID * 160  + 255) / 256, 256, 0, stream>>>(ew1, W1t, HID, CIN, 160);
    pack_w3x3_bf16<<<(HID * 1152 + 255) / 256, 256, 0, stream>>>(ew2, W2t, HID, HID, 1152);
    pack_w3x3_bf16<<<(HID * 1152 + 255) / 256, 256, 0, stream>>>(dw1, W3t, HID, HID, 1152);
    cast_f32_to_bf16<<<(CIN * HID + 255) / 256, 256, 0, stream>>>(dw2, W4t, CIN * HID);
    cast_f32_to_bf16<<<(KCODES * HID + 255) / 256, 256, 0, stream>>>(embed, E, KCODES * HID);
    embed_norm2<<<(KCODES + 255) / 256, 256, 0, stream>>>(embed, N2);

    // ---- encoder --------------------------------------------------------
    conv3x3_wmma<<<MTOT / 16, 256, 0, stream>>>(X0, W1t, eb1, H1, CIN, 160);
    conv3x3_wmma<<<MTOT / 16, 256, 0, stream>>>(H1, W2t, eb2, H2, HID, 1152);

    // ---- vector quantize ------------------------------------------------
    vq_argmin_gather<<<MTOT / 16, 256, 0, stream>>>(H2, E, N2, Q);

    // ---- decoder --------------------------------------------------------
    conv3x3_wmma<<<MTOT / 16, 256, 0, stream>>>(Q, W3t, db1, Y3, HID, 1152);
    head1x1_wmma<<<MTOT / 128, 256, 0, stream>>>(Y3, W4t, db2, out);
}
